// MPAttention_39384850104938
// MI455X (gfx1250) — compile-verified, matrix operand order checked by reading, not run on hardware
//
#include <hip/hip_runtime.h>
#include <hip/hip_bf16.h>

#define B_     2
#define N_     2048
#define H_     16
#define D_     64
#define DIM_   1024
#define TRIPLE_ 3072
#define NKV_   2052
#define NKVP_  2080   // padded to multiple of 32

typedef __attribute__((ext_vector_type(16))) __bf16 v16bf;
typedef __attribute__((ext_vector_type(8)))  float  v8f;

union FragU { uint4 u[2]; v16bf v; };

#define WMMA_BF16(a, b, c) \
  __builtin_amdgcn_wmma_f32_16x16x32_bf16(false, (a), false, (b), (short)0, (c), false, false)

#if defined(__gfx1250__) && __has_builtin(__builtin_amdgcn_global_load_async_to_lds_b128)
#define ASYNC_CP 1
typedef int v4i_async __attribute__((vector_size(16)));
typedef __attribute__((address_space(1))) v4i_async* async_gptr_t;
typedef __attribute__((address_space(3))) v4i_async* async_lptr_t;
#else
#define ASYNC_CP 0
#endif

__device__ __forceinline__ unsigned short f2bf(float f) {
  unsigned int u = __builtin_bit_cast(unsigned int, f);
  u += 0x7FFFu + ((u >> 16) & 1u);            // round-to-nearest-even
  return (unsigned short)(u >> 16);
}

__device__ __forceinline__ float wave_sum(float s) {
#pragma unroll
  for (int m = 1; m < 32; m <<= 1) s += __shfl_xor(s, m, 32);
  return s;
}

// One b128 worth of global->LDS copy: async (ASYNCcnt-tracked) when available.
__device__ __forceinline__ void cp_b128(const unsigned short* g, unsigned short* l) {
#if ASYNC_CP
  __builtin_amdgcn_global_load_async_to_lds_b128(
      (async_gptr_t)(g), (async_lptr_t)(l), 0, 0);
#else
  *(uint4*)l = *(const uint4*)g;
#endif
}

template <int Np>
__device__ __forceinline__ void wait_async() {
#if defined(__gfx1250__)
#if __has_builtin(__builtin_amdgcn_s_wait_asynccnt)
  __builtin_amdgcn_s_wait_asynccnt(Np);
#else
  asm volatile("s_wait_asynccnt %0" ::"i"(Np) : "memory");
#endif
#endif
}

// Load one 16x32 bf16 fragment (A layout; B is loaded as B^T rows with the
// same pattern). Per ISA 16-bit A layout: lane L holds row M=L%16,
// K = {0..7,16..23} (lanes<16) or {8..15,24..31} (lanes>=16): two b128 loads.
__device__ __forceinline__ v16bf load_frag(const unsigned short* tile, int stride) {
  int lane = threadIdx.x & 31;
  const unsigned short* p = tile + (size_t)(lane & 15) * stride + ((lane >> 4) << 3);
  FragU f;
  f.u[0] = *(const uint4*)(p);
  f.u[1] = *(const uint4*)(p + 16);
  return f.v;
}

// ---------------- shared GEMM core: C[128x64] per block, K=1024 ------------
// A  : [Mtot, 1024] bf16 row-major (activations)
// Bt : [Ntot, 1024] bf16 row-major (normalized weights; B^T of the GEMM)
// LDS double-buffered 128x32 A-tile + 64x32 B-tile staged with async copies.
__device__ __forceinline__ void gemm_core(const unsigned short* __restrict__ A,
                                          const unsigned short* __restrict__ Bt,
                                          int m0blk, int n0,
                                          unsigned short* As, unsigned short* Bs,
                                          v8f acc[4]) {
  const int K = DIM_;
  int tid = threadIdx.x;
  int wid = tid >> 5;

  auto stage = [&](int buf, int k0) {
    unsigned short* abuf = As + buf * (128 * 32);
    unsigned short* bbuf = Bs + buf * (64 * 32);
#pragma unroll
    for (int i = 0; i < 2; ++i) {                 // A tile: 512 b128 chunks
      int c = tid + i * 256;
      int row = c >> 2, col = (c & 3) << 3;
      cp_b128(A + (size_t)(m0blk + row) * K + k0 + col, abuf + row * 32 + col);
    }
    {                                             // B tile: 256 b128 chunks
      int row = tid >> 2, col = (tid & 3) << 3;
      cp_b128(Bt + (size_t)(n0 + row) * K + k0 + col, bbuf + row * 32 + col);
    }
  };

  stage(0, 0);
  for (int k0 = 0; k0 < K; k0 += 32) {
    int cur = (k0 >> 5) & 1;
    if (k0 + 32 < K) {
      stage(cur ^ 1, k0 + 32);   // overlap next stage with this compute
      wait_async<3>();           // previous batch (this buffer) landed
    } else {
      wait_async<0>();
    }
    __syncthreads();
    const unsigned short* abuf = As + cur * (128 * 32) + wid * 16 * 32;
    const unsigned short* bbuf = Bs + cur * (64 * 32);
    v16bf a = load_frag(abuf, 32);
#pragma unroll
    for (int t = 0; t < 4; ++t) {
      v16bf b = load_frag(bbuf + t * 16 * 32, 32);
      acc[t] = WMMA_BF16(a, b, acc[t]);
    }
    __syncthreads();             // all waves done reading before overwrite
  }
}

// ---------------- weight normalization: l2norm(rows)/sqrt(1024) -> bf16 ----
__global__ void wnorm_kernel(const float* __restrict__ wqkv, const float* __restrict__ wout,
                             unsigned short* __restrict__ wnq, unsigned short* __restrict__ wno) {
  __shared__ float part[8];
  int row = blockIdx.x;
  const float* src;
  unsigned short* dst;
  if (row < TRIPLE_) { src = wqkv + (size_t)row * DIM_;            dst = wnq + (size_t)row * DIM_; }
  else               { src = wout + (size_t)(row - TRIPLE_) * DIM_; dst = wno + (size_t)(row - TRIPLE_) * DIM_; }
  float v[4];
  float s = 0.f;
#pragma unroll
  for (int i = 0; i < 4; ++i) { v[i] = src[threadIdx.x + i * 256]; s += v[i] * v[i]; }
  s = wave_sum(s);
  if ((threadIdx.x & 31) == 0) part[threadIdx.x >> 5] = s;
  __syncthreads();
  float tot = 0.f;
#pragma unroll
  for (int i = 0; i < 8; ++i) tot += part[i];
  float inv = 1.0f / (fmaxf(sqrtf(tot), 1e-4f) * 32.0f);   // /sqrt(fan_in=1024)
#pragma unroll
  for (int i = 0; i < 4; ++i) dst[threadIdx.x + i * 256] = f2bf(v[i] * inv);
}

// ---------------- x -> bf16 ------------------------------------------------
__global__ void xcast_kernel(const float* __restrict__ x, unsigned short* __restrict__ xb, int n) {
  int i = blockIdx.x * blockDim.x + threadIdx.x;
  if (i < n) xb[i] = f2bf(x[i]);
}

// ---------------- QKV GEMM: [4096,1024] x [1024,3072] (Wn^T) ---------------
__global__ void gemm_qkv_kernel(const unsigned short* __restrict__ xb,
                                const unsigned short* __restrict__ wn,
                                float* __restrict__ qkv) {
  __shared__ unsigned short As[2 * 128 * 32];
  __shared__ unsigned short Bs[2 * 64 * 32];
  int wid  = threadIdx.x >> 5;
  int lane = threadIdx.x & 31;
  int m0blk = blockIdx.y * 128;
  int n0    = blockIdx.x * 64;
  v8f acc[4] = {};
  gemm_core(xb, wn, m0blk, n0, As, Bs, acc);
  // C layout: element (M = r + 8*(lane>>4), N = lane&15)
  int mb = m0blk + wid * 16 + ((lane >> 4) << 3);
  int nb = n0 + (lane & 15);
#pragma unroll
  for (int t = 0; t < 4; ++t)
#pragma unroll
    for (int r = 0; r < 8; ++r)
      qkv[(size_t)(mb + r) * TRIPLE_ + nb + t * 16] = acc[t][r];
}

// ---------------- pixel-norm of q/k/v, pack bf16, build K and V^T ----------
__global__ void qkvnorm_kernel(const float* __restrict__ qkv,
                               unsigned short* __restrict__ qb,
                               unsigned short* __restrict__ kb,
                               unsigned short* __restrict__ vt) {
  int gw   = (blockIdx.x * blockDim.x + threadIdx.x) >> 5;  // one wave per 64-vec
  int lane = threadIdx.x & 31;
  int h   = gw % H_;
  int sel = (gw / H_) % 3;
  int n   = (gw / (3 * H_)) % N_;
  int b   = gw / (3 * H_ * N_);
  const float* src = qkv + (size_t)(b * N_ + n) * TRIPLE_ + sel * DIM_ + h * D_;
  float2 e = *(const float2*)(src + lane * 2);
  float s = wave_sum(e.x * e.x + e.y * e.y);
  float inv = 1.0f / fmaxf(sqrtf(s), 1e-4f);
  // q: l2norm*sqrt(d)*d^-0.5 = l2norm ; k,v: l2norm*sqrt(d) = 8*l2norm
  if (sel != 0) inv *= 8.0f;
  int bh = b * H_ + h;
  if (sel == 0) {
    unsigned int pk = (unsigned int)f2bf(e.x * inv) | ((unsigned int)f2bf(e.y * inv) << 16);
    *(unsigned int*)(qb + ((size_t)bh * N_ + n) * D_ + lane * 2) = pk;
  } else if (sel == 1) {
    unsigned int pk = (unsigned int)f2bf(e.x * inv) | ((unsigned int)f2bf(e.y * inv) << 16);
    *(unsigned int*)(kb + ((size_t)bh * NKVP_ + 4 + n) * D_ + lane * 2) = pk;
  } else {
    vt[((size_t)bh * D_ + lane * 2)     * NKVP_ + 4 + n] = f2bf(e.x * inv);
    vt[((size_t)bh * D_ + lane * 2 + 1) * NKVP_ + 4 + n] = f2bf(e.y * inv);
  }
}

// ---------------- mem_kv -> kv slots 0..3 (broadcast over batch) -----------
__global__ void memkv_kernel(const float* __restrict__ mem,
                             unsigned short* __restrict__ kb,
                             unsigned short* __restrict__ vt) {
  int lane = threadIdx.x & 31;
  int id = blockIdx.x;
  int s = id & 3;
  int h = (id >> 2) % H_;
  int b = id / (4 * H_);
  int bh = b * H_ + h;
  {  // K memory slot
    const float* mk = mem + ((size_t)(0 * H_ + h) * 4 + s) * D_;
    float2 e = *(const float2*)(mk + lane * 2);
    float ss = wave_sum(e.x * e.x + e.y * e.y);
    float inv = 8.0f / fmaxf(sqrtf(ss), 1e-4f);
    unsigned int pk = (unsigned int)f2bf(e.x * inv) | ((unsigned int)f2bf(e.y * inv) << 16);
    *(unsigned int*)(kb + ((size_t)bh * NKVP_ + s) * D_ + lane * 2) = pk;
  }
  {  // V memory slot (into V^T)
    const float* mv = mem + ((size_t)(1 * H_ + h) * 4 + s) * D_;
    float2 e = *(const float2*)(mv + lane * 2);
    float ss = wave_sum(e.x * e.x + e.y * e.y);
    float inv = 8.0f / fmaxf(sqrtf(ss), 1e-4f);
    vt[((size_t)bh * D_ + lane * 2)     * NKVP_ + s] = f2bf(e.x * inv);
    vt[((size_t)bh * D_ + lane * 2 + 1) * NKVP_ + s] = f2bf(e.y * inv);
  }
}

// ---------------- zero kv padding region [2052, 2080) ----------------------
__global__ void pad_kernel(unsigned short* __restrict__ kb, unsigned short* __restrict__ vt) {
  int bh = blockIdx.x;
  for (int i = threadIdx.x; i < (NKVP_ - NKV_) * D_; i += blockDim.x) {
    int slot = NKV_ + i / D_;
    int d    = i % D_;
    kb[((size_t)bh * NKVP_ + slot) * D_ + d] = 0;
    vt[((size_t)bh * D_ + d) * NKVP_ + slot] = 0;
  }
}

// ---------------- flash attention: 1 wave per 16-query tile ----------------
__global__ void attn_kernel(const unsigned short* __restrict__ qb,
                            const unsigned short* __restrict__ kb,
                            const unsigned short* __restrict__ vt,
                            unsigned short* __restrict__ ao) {
  __shared__ unsigned short P[16 * 32];   // prob tile staged for A-frag reload
  int lane = threadIdx.x & 31;
  int blk = blockIdx.x;
  int qt = blk & 127;
  int h  = (blk >> 7) & 15;
  int b  = blk >> 11;
  int bh = b * H_ + h;
  const unsigned short* qtile = qb + ((size_t)bh * N_ + qt * 16) * D_;
  v16bf qa0 = load_frag(qtile, D_);
  v16bf qa1 = load_frag(qtile + 32, D_);
  const unsigned short* kbase = kb + (size_t)bh * NKVP_ * D_;
  const unsigned short* vbase = vt + (size_t)bh * D_ * NKVP_;
  v8f o[4] = {};
  float mrow[8], lrow[8];
#pragma unroll
  for (int r = 0; r < 8; ++r) { mrow[r] = -1e30f; lrow[r] = 0.f; }
  int rowb = (lane >> 4) << 3;

  for (int kv = 0; kv < NKVP_; kv += 32) {
    v8f s0 = {}, s1 = {};
    s0 = WMMA_BF16(qa0, load_frag(kbase + (size_t)kv * D_,            D_), s0);
    s0 = WMMA_BF16(qa1, load_frag(kbase + (size_t)kv * D_ + 32,       D_), s0);
    s1 = WMMA_BF16(qa0, load_frag(kbase + (size_t)(kv + 16) * D_,     D_), s1);
    s1 = WMMA_BF16(qa1, load_frag(kbase + (size_t)(kv + 16) * D_ + 32, D_), s1);
    int c0 = kv + (lane & 15);
    if (c0 >= NKV_) {
#pragma unroll
      for (int r = 0; r < 8; ++r) s0[r] = -1e30f;
    }
    if (c0 + 16 >= NKV_) {
#pragma unroll
      for (int r = 0; r < 8; ++r) s1[r] = -1e30f;
    }
#pragma unroll
    for (int r = 0; r < 8; ++r) {
      float tmax = fmaxf(s0[r], s1[r]);
#pragma unroll
      for (int m = 1; m < 16; m <<= 1) tmax = fmaxf(tmax, __shfl_xor(tmax, m, 32));
      float mn   = fmaxf(mrow[r], tmax);
      float corr = __expf(mrow[r] - mn);
      float p0 = __expf(s0[r] - mn);
      float p1 = __expf(s1[r] - mn);
      float rs = p0 + p1;
#pragma unroll
      for (int m = 1; m < 16; m <<= 1) rs += __shfl_xor(rs, m, 32);
      lrow[r] = lrow[r] * corr + rs;
      mrow[r] = mn;
      o[0][r] *= corr; o[1][r] *= corr; o[2][r] *= corr; o[3][r] *= corr;
      P[(rowb + r) * 32 + (lane & 15)]      = f2bf(p0);
      P[(rowb + r) * 32 + 16 + (lane & 15)] = f2bf(p1);
    }
    __syncthreads();
    v16bf pa = load_frag(P, 32);   // LDS reload converts C-layout -> A-layout
#pragma unroll
    for (int t = 0; t < 4; ++t) {
      v16bf bv = load_frag(vbase + (size_t)(t * 16) * NKVP_ + kv, NKVP_);
      o[t] = WMMA_BF16(pa, bv, o[t]);
    }
    __syncthreads();
  }
#pragma unroll
  for (int r = 0; r < 8; ++r) {
    float invl = 1.0f / fmaxf(lrow[r], 1e-20f);
    int token = qt * 16 + rowb + r;
    size_t base = ((size_t)(b * N_ + token) * H_ + h) * D_;
#pragma unroll
    for (int t = 0; t < 4; ++t)
      ao[base + t * 16 + (lane & 15)] = f2bf(o[t][r] * invl);
  }
}

// ---------------- out projection + magnitude-preserving residual -----------
__global__ void gemm_out_kernel(const unsigned short* __restrict__ ao,
                                const unsigned short* __restrict__ wno,
                                const float* __restrict__ x,
                                float* __restrict__ out) {
  __shared__ unsigned short As[2 * 128 * 32];
  __shared__ unsigned short Bs[2 * 64 * 32];
  int wid  = threadIdx.x >> 5;
  int lane = threadIdx.x & 31;
  int m0blk = blockIdx.y * 128;
  int n0    = blockIdx.x * 64;
  v8f acc[4] = {};
  gemm_core(ao, wno, m0blk, n0, As, Bs, acc);
  const float MP_SCALE = 1.3130643286f;   // 1/sqrt(0.7^2 + 0.3^2)
  int mb = m0blk + wid * 16 + ((lane >> 4) << 3);
  int nb = n0 + (lane & 15);
#pragma unroll
  for (int t = 0; t < 4; ++t)
#pragma unroll
    for (int r = 0; r < 8; ++r) {
      size_t idx = (size_t)(mb + r) * DIM_ + nb + t * 16;
      out[idx] = (acc[t][r] * 0.7f + x[idx] * 0.3f) * MP_SCALE;
    }
}

extern "C" void kernel_launch(void* const* d_in, const int* in_sizes, int n_in,
                              void* d_out, int out_size, void* d_ws, size_t ws_size,
                              hipStream_t stream) {
  (void)in_sizes; (void)n_in; (void)out_size; (void)ws_size;
  const float* x     = (const float*)d_in[0];
  const float* wqkv  = (const float*)d_in[1];
  const float* wout  = (const float*)d_in[2];
  const float* memkv = (const float*)d_in[3];
  float* out = (float*)d_out;

  char* ws = (char*)d_ws;
  size_t off = 0;
  auto alloc = [&](size_t bytes) {
    char* p = ws + off;
    off += (bytes + 255) & ~(size_t)255;
    return p;
  };
  unsigned short* wnq  = (unsigned short*)alloc((size_t)TRIPLE_ * DIM_ * 2);
  unsigned short* wno  = (unsigned short*)alloc((size_t)DIM_ * DIM_ * 2);
  unsigned short* xb   = (unsigned short*)alloc((size_t)B_ * N_ * DIM_ * 2);
  float*          qkvf = (float*)         alloc((size_t)B_ * N_ * TRIPLE_ * 4);
  unsigned short* qbuf = (unsigned short*)alloc((size_t)B_ * H_ * N_ * D_ * 2);
  unsigned short* kbuf = (unsigned short*)alloc((size_t)B_ * H_ * NKVP_ * D_ * 2);
  unsigned short* vtb  = (unsigned short*)alloc((size_t)B_ * H_ * D_ * NKVP_ * 2);
  unsigned short* ao   = (unsigned short*)alloc((size_t)B_ * N_ * H_ * D_ * 2);

  wnorm_kernel<<<TRIPLE_ + DIM_, 256, 0, stream>>>(wqkv, wout, wnq, wno);
  int xtot = B_ * N_ * DIM_;
  xcast_kernel<<<(xtot + 255) / 256, 256, 0, stream>>>(x, xb, xtot);
  gemm_qkv_kernel<<<dim3(TRIPLE_ / 64, (B_ * N_) / 128), 256, 0, stream>>>(xb, wnq, qkvf);
  qkvnorm_kernel<<<(B_ * N_ * 3 * H_) / 8, 256, 0, stream>>>(qkvf, qbuf, kbuf, vtb);
  memkv_kernel<<<B_ * H_ * 4, 32, 0, stream>>>(memkv, kbuf, vtb);
  pad_kernel<<<B_ * H_, 256, 0, stream>>>(kbuf, vtb);
  attn_kernel<<<B_ * H_ * (N_ / 16), 32, 0, stream>>>(qbuf, kbuf, vtb, ao);
  gemm_out_kernel<<<dim3(DIM_ / 64, (B_ * N_) / 128), 256, 0, stream>>>(ao, wno, x, out);
}
